// MRSDS_32169305047653
// MI455X (gfx1250) — matrix operand order repaired and checked by reading, not compiled, not run on hardware
//
#include <hip/hip_runtime.h>
#include <hip/hip_bf16.h>
#include <cstdint>
#include <cstddef>

// ---------------------------------------------------------------------------
// MRSDS ELBO on MI455X (gfx1250).  All GEMMs via v_wmma_f32_16x16x32_bf16
// (wave32 WMMA, f32 accumulate).  Reductions / logsumexp / scan in f32.
// ---------------------------------------------------------------------------

#define LOG2PI_F 1.8378770664093453f

constexpr int Bv = 32, Tv = 1024, Dv = 512, Hv = 512, Xv = 32, Kv = 8;
constexpr int Nv = Bv * Tv;              // 32768 rows

typedef __attribute__((ext_vector_type(16))) __bf16 bfvec16;
typedef __attribute__((ext_vector_type(8)))  float  f32x8;

static __device__ inline f32x8 wmma_bf16f32(bfvec16 a, bfvec16 b, f32x8 c) {
  // D = A(16x32 bf16) * B(32x16 bf16) + C(16x16 f32)
  return __builtin_amdgcn_wmma_f32_16x16x32_bf16(false, a, false, b,
                                                 (short)0, c, false, false);
}

// 16-bit A-fragment slot map (16x32 tile): element (row M, col K) lives at
// lane = half*16 + M, byte-slot e, per CDNA5 ISA 7.12.2.
//   K 0..7  -> half0,e=K     ; K 8..15  -> half1,e=K-8
//   K 16..23-> half0,e=K-8   ; K 24..31 -> half1,e=K-16
static __device__ inline void afrag_slot(int colK, int rowM, int& lane, int& e) {
  int half = (colK >> 3) & 1;
  e = (colK & 7) | ((colK & 16) ? 8 : 0);
  lane = half * 16 + rowM;
}

// ---------------------------------------------------------------------------
// Workspace layout (bytes).  Packed weights first, then f32 intermediates.
// ---------------------------------------------------------------------------
constexpr size_t SZ_W1B  = (size_t)(Hv / 16) * (Dv / 32) * 512 * 2; // 512 KB
constexpr size_t SZ_WXB  = (size_t)(Xv / 16) * (Dv / 32) * 512 * 2; //  32 KB
constexpr size_t SZ_WZB  = (size_t)((Kv * Kv) / 16) * (Xv / 32) * 512 * 2;
constexpr size_t SZ_CB   = (size_t)(Dv / 16) * (Xv / 32) * 512 * 2;
constexpr size_t SZ_AB   = (size_t)Kv * (Xv / 16) * (Xv / 32) * 512 * 2;

constexpr size_t OFF_W1B   = 0;
constexpr size_t OFF_WMUB  = OFF_W1B + SZ_W1B;
constexpr size_t OFF_WLVB  = OFF_WMUB + SZ_WXB;
constexpr size_t OFF_WZB   = OFF_WLVB + SZ_WXB;
constexpr size_t OFF_CB    = OFF_WZB + SZ_WZB;
constexpr size_t OFF_AB    = OFF_CB + SZ_CB;
constexpr size_t OFF_X     = OFF_AB + SZ_AB;                       // N*X f32
constexpr size_t OFF_ENT   = OFF_X + (size_t)Nv * Xv * 4;          // N f32
constexpr size_t OFF_LOGA  = OFF_ENT + (size_t)Nv * 4;             // N*K*K f32
constexpr size_t OFF_LOGB  = OFF_LOGA + (size_t)Nv * Kv * Kv * 4;  // N*K f32
constexpr size_t OFF_MEANS = OFF_LOGB + (size_t)Nv * Kv * 4;       // N*K*X f32
constexpr size_t OFF_FWD   = OFF_MEANS + (size_t)Nv * Kv * Xv * 4; // T*B*K f32
constexpr size_t OFF_BWD   = OFF_FWD + (size_t)Tv * Bv * Kv * 4;
constexpr size_t OFF_LPX   = OFF_BWD + (size_t)Tv * Bv * Kv * 4;   // B f32
constexpr size_t OFF_LPY   = OFF_LPX + 256;                        // N f32
// total ~48 MB

// ---------------------------------------------------------------------------
// Pack f32 weight matrix [nMat][Kd x Nd] (row-major, K rows) into WMMA
// B-fragment bf16 layout: P[((m*nN + nt)*nK + kc)*32 + lane][e], where
// B element (K,N): K = kc*32 + (lane/16)*8 + (e&7) + (e>>3)*16, N = nt*16 + lane%16.
// ---------------------------------------------------------------------------
__global__ __launch_bounds__(256) void pack_b_kernel(
    const float* __restrict__ W, __bf16* __restrict__ P,
    int Kd, int Nd, int nMat) {
  int idx = blockIdx.x * blockDim.x + threadIdx.x;
  int nK = Kd >> 5, nN = Nd >> 4;
  int per = nN * nK * 512;
  if (idx >= per * nMat) return;
  int m = idx / per, r = idx - m * per;
  int e = r & 15, lane = (r >> 4) & 31;
  int kc = (r >> 9) % nK, nt = (r >> 9) / nK;
  int k = kc * 32 + ((lane >> 4) << 3) + (e & 7) + ((e >> 3) << 4);
  int n = nt * 16 + (lane & 15);
  P[idx] = (__bf16)W[(size_t)m * Kd * Nd + (size_t)k * Nd + n];
}

// ---------------------------------------------------------------------------
// Kernel A: fused inference net.  One block = 16 rows of [N, D].
// 8 waves; stage1: h = tanh(ys@W1+b1) (each wave owns 64 H-cols, 16 k-chunks)
// stage2: mu/lv = h@{Wmu,Wlv} (k-chunks split across waves, LDS reduce)
// stage3: x = mu + exp(.5 lv)*eps, per-row entropy.
// ---------------------------------------------------------------------------
__global__ __launch_bounds__(256) void infer_kernel(
    const float* __restrict__ ys, const float* __restrict__ eps,
    const __bf16* __restrict__ W1b, const float* __restrict__ b1,
    const __bf16* __restrict__ Wmub, const float* __restrict__ bmu,
    const __bf16* __restrict__ Wlvb, const float* __restrict__ blv,
    float* __restrict__ x_out, float* __restrict__ ent_out) {
  __shared__ alignas(32) __bf16 aFrag[16][32][16];  // ys tile (A-frag layout)
  __shared__ alignas(32) __bf16 hFrag[16][32][16];  // h  tile (A-frag layout)
  __shared__ float muv[16][32];
  __shared__ float lvv[16][32];
  __shared__ float entbuf[16][32];

  const int tid = threadIdx.x;
  const int lane = tid & 31, wave = tid >> 5;
  const int r0 = blockIdx.x * 16;

  // zero partial-sum buffers
  for (int it = tid; it < 512; it += 256) {
    ((float*)muv)[it] = 0.f;
    ((float*)lvv)[it] = 0.f;
  }
  // stage-in ys tile -> bf16 A-fragment layout
  for (int idx = tid; idx < 16 * Dv; idx += 256) {
    int row = idx >> 9, col = idx & 511;
    float v = ys[(size_t)(r0 + row) * Dv + col];
    int kc = col >> 5, kk = col & 31, ln, e;
    afrag_slot(kk, row, ln, e);
    aFrag[kc][ln][e] = (__bf16)v;
  }
  __syncthreads();

  const int mrow = lane & 15;     // N within 16-wide tile for C fragments
  const int hi = lane >> 4;       // row-half select

  // ---- stage 1: h = tanh(ys @ W1 + b1), wave owns H-cols [64w, 64w+64) ----
  f32x8 zero8 = {0.f, 0.f, 0.f, 0.f, 0.f, 0.f, 0.f, 0.f};
  f32x8 acc[4] = {zero8, zero8, zero8, zero8};
  for (int kc = 0; kc < 16; ++kc) {
    bfvec16 a = *(const bfvec16*)&aFrag[kc][lane][0];
#pragma unroll
    for (int i = 0; i < 4; ++i) {
      int nt = wave * 4 + i;
      const __bf16* bp = W1b + (((size_t)nt * 16 + kc) * 32 + lane) * 16;
      if (kc + 1 < 16)
        __builtin_prefetch(W1b + (((size_t)nt * 16 + kc + 1) * 32 + lane) * 16, 0, 3);
      bfvec16 bv = *(const bfvec16*)bp;
      acc[i] = wmma_bf16f32(a, bv, acc[i]);
    }
  }
#pragma unroll
  for (int i = 0; i < 4; ++i) {
    int col = wave * 64 + i * 16 + mrow;       // h column (stage-2 K index)
    float bb = b1[col];
    int kc2 = col >> 5, kk = col & 31, ln, e;
    afrag_slot(kk, 0, ln, e);                  // row filled below
#pragma unroll
    for (int r = 0; r < 8; ++r) {
      int row = r + hi * 8;
      float hv = tanhf(acc[i][r] + bb);
      hFrag[kc2][(ln & 16) + row][e] = (__bf16)hv;
    }
  }
  __syncthreads();

  // ---- stage 2: mu/lv = h @ Wmu / Wlv ; wave handles k-chunks {2w,2w+1} ---
  f32x8 amu[2] = {zero8, zero8};
  f32x8 alv[2] = {zero8, zero8};
#pragma unroll
  for (int kk2 = 0; kk2 < 2; ++kk2) {
    int kc = wave * 2 + kk2;
    bfvec16 a = *(const bfvec16*)&hFrag[kc][lane][0];
#pragma unroll
    for (int i = 0; i < 2; ++i) {
      bfvec16 bm = *(const bfvec16*)(Wmub + (((size_t)i * 16 + kc) * 32 + lane) * 16);
      bfvec16 bl = *(const bfvec16*)(Wlvb + (((size_t)i * 16 + kc) * 32 + lane) * 16);
      amu[i] = wmma_bf16f32(a, bm, amu[i]);
      alv[i] = wmma_bf16f32(a, bl, alv[i]);
    }
  }
#pragma unroll
  for (int i = 0; i < 2; ++i)
#pragma unroll
    for (int r = 0; r < 8; ++r) {
      atomicAdd(&muv[r + hi * 8][i * 16 + mrow], amu[i][r]);
      atomicAdd(&lvv[r + hi * 8][i * 16 + mrow], alv[i][r]);
    }
  __syncthreads();

  // ---- stage 3: sample + entropy ----
  for (int item = tid; item < 512; item += 256) {
    int row = item >> 5, col = item & 31;
    float mu = muv[row][col] + bmu[col];
    float lv = lvv[row][col] + blv[col];
    size_t g = (size_t)(r0 + row) * Xv + col;
    float ev = eps[g];
    x_out[g] = mu + __expf(0.5f * lv) * ev;
    entbuf[row][col] = 0.5f * (ev * ev + lv + LOG2PI_F);
  }
  __syncthreads();
  if (tid < 16) {
    float s = 0.f;
    for (int c = 0; c < 32; ++c) s += entbuf[tid][c];
    ent_out[r0 + tid] = s;
  }
}

// ---------------------------------------------------------------------------
// Kernel B: dynamics.  One block = 16 time steps of one batch.
// logits = x_prev@Wz+bz (log-softmax over j); means[k] = x_prev@A[k]+u@Bu[k]+bx;
// log_b[t+1,k] = sum_x gauss_lp(x_next, means[k], x_logvar[k]).
// ---------------------------------------------------------------------------
__global__ __launch_bounds__(256) void dyn_kernel(
    const float* __restrict__ x, const float* __restrict__ us,
    const __bf16* __restrict__ Wzb, const float* __restrict__ bz,
    const __bf16* __restrict__ Ab, const float* __restrict__ Bu,
    const float* __restrict__ bx, const float* __restrict__ x_logvar,
    float* __restrict__ log_a, float* __restrict__ log_b,
    float* __restrict__ means_out) {
  __shared__ alignas(32) __bf16 xF[32][16];  // prev-x tile, A-frag layout
  __shared__ float xn[16][32];
  __shared__ float mns[16][8][32];
  __shared__ float lgt[16][64];

  const int tid = threadIdx.x;
  const int lane = tid & 31, wave = tid >> 5;
  const int bidx = blockIdx.x >> 6;     // batch
  const int t0 = (blockIdx.x & 63) * 16;

  for (int it = tid; it < 512; it += 256) {
    int row = it >> 5, col = it & 31;
    int t = t0 + row;
    float vp = x[((size_t)bidx * Tv + t) * Xv + col];
    int ln, e;
    afrag_slot(col, row, ln, e);
    xF[ln][e] = (__bf16)vp;
    int tn = (t + 1 < Tv) ? t + 1 : Tv - 1;   // t==1023 rows are masked later
    xn[row][col] = x[((size_t)bidx * Tv + tn) * Xv + col];
  }
  __syncthreads();

  const int mrow = lane & 15, hi = lane >> 4;
  f32x8 zero8 = {0.f, 0.f, 0.f, 0.f, 0.f, 0.f, 0.f, 0.f};
  bfvec16 a = *(const bfvec16*)&xF[lane][0];

  // means for state k == wave (2 n-tiles of X=32)
  f32x8 am[2] = {zero8, zero8};
#pragma unroll
  for (int i = 0; i < 2; ++i) {
    bfvec16 bv = *(const bfvec16*)(Ab + (((size_t)wave * 2 + i) * 32 + lane) * 16);
    am[i] = wmma_bf16f32(a, bv, am[i]);
  }
#pragma unroll
  for (int i = 0; i < 2; ++i)
#pragma unroll
    for (int r = 0; r < 8; ++r)
      mns[r + hi * 8][wave][i * 16 + mrow] = am[i][r];

  if (wave < 4) {   // logits n-tile per wave (K*K = 64 cols); wave-uniform
    f32x8 al = zero8;
    bfvec16 bv = *(const bfvec16*)(Wzb + ((size_t)wave * 32 + lane) * 16);
    al = wmma_bf16f32(a, bv, al);
    int col = wave * 16 + mrow;
    float bzv = bz[col];
#pragma unroll
    for (int r = 0; r < 8; ++r)
      lgt[r + hi * 8][col] = al[r] + bzv;
  }
  __syncthreads();

  if (tid < 128) {
    int row = tid >> 3, k = tid & 7;
    int t = t0 + row;
    bool valid = (t < Tv - 1);
    size_t gp = (size_t)bidx * Tv + t;
    int tn = valid ? t + 1 : Tv - 1;
    float u0 = us[((size_t)bidx * Tv + tn) * 2 + 0];
    float u1 = us[((size_t)bidx * Tv + tn) * 2 + 1];
    float lb = 0.f;
    for (int c = 0; c < Xv; ++c) {
      float m = mns[row][k][c] + u0 * Bu[(k * 2 + 0) * Xv + c]
              + u1 * Bu[(k * 2 + 1) * Xv + c] + bx[k * Xv + c];
      float xlv = x_logvar[k * Xv + c];
      float df = xn[row][c] - m;
      lb += -0.5f * (df * df * __expf(-xlv) + xlv + LOG2PI_F);
      if (valid) means_out[(gp * Kv + k) * Xv + c] = m;
    }
    if (valid) log_b[((size_t)bidx * Tv + t + 1) * Kv + k] = lb;

    // column log-softmax over j (axis=-2) for column k
    float mx = -1e30f;
    for (int j = 0; j < Kv; ++j) mx = fmaxf(mx, lgt[row][j * Kv + k]);
    float s = 0.f;
    for (int j = 0; j < Kv; ++j) s += __expf(lgt[row][j * Kv + k] - mx);
    float lse = mx + __logf(s);
    if (valid)
      for (int j = 0; j < Kv; ++j)
        log_a[((size_t)bidx * Tv + t + 1) * 64 + j * Kv + k] =
            lgt[row][j * Kv + k] - lse;
  }
}

// ---------------------------------------------------------------------------
// Kernel C: log_b at t=0 from x0 prior.
// ---------------------------------------------------------------------------
__global__ __launch_bounds__(256) void logb0_kernel(
    const float* __restrict__ x, const float* __restrict__ x0_mean,
    const float* __restrict__ x0_logvar, float* __restrict__ log_b) {
  int tid = threadIdx.x;
  int b = tid >> 3, k = tid & 7;
  float lb = 0.f;
  for (int c = 0; c < Xv; ++c) {
    float xv = x[((size_t)b * Tv) * Xv + c];
    float m = x0_mean[k * Xv + c], lv = x0_logvar[k * Xv + c];
    float d = xv - m;
    lb += -0.5f * (d * d * __expf(-lv) + lv + LOG2PI_F);
  }
  log_b[((size_t)b * Tv) * Kv + k] = lb;
}

// ---------------------------------------------------------------------------
// Kernel D: forward-backward scan.  Single workgroup, 256 threads = (b, j).
// alpha/beta live in LDS; 1023 sequential steps each direction.
// ---------------------------------------------------------------------------
__global__ __launch_bounds__(256) void fb_kernel(
    const float* __restrict__ log_a, const float* __restrict__ log_b,
    const float* __restrict__ log_init_z, float* __restrict__ fwd,
    float* __restrict__ bwd, float* __restrict__ log_px) {
  __shared__ float alpha[32][8];
  __shared__ float tmp[32][8];
  __shared__ float lpx[32];
  const int tid = threadIdx.x;
  const int b = tid >> 3, j = tid & 7;

  float mz = -1e30f;
  for (int i = 0; i < Kv; ++i) mz = fmaxf(mz, log_init_z[i]);
  float sz = 0.f;
  for (int i = 0; i < Kv; ++i) sz += __expf(log_init_z[i] - mz);
  float linit = log_init_z[j] - (mz + __logf(sz));

  float a0 = linit + log_b[((size_t)b * Tv) * Kv + j];
  tmp[b][j] = a0;
  __syncthreads();
  float m = -1e30f;
  for (int i = 0; i < Kv; ++i) m = fmaxf(m, tmp[b][i]);
  float s = 0.f;
  for (int i = 0; i < Kv; ++i) s += __expf(tmp[b][i] - m);
  float c0 = m + __logf(s);
  alpha[b][j] = a0 - c0;
  if (j == 0) lpx[b] = c0;
  fwd[b * Kv + j] = a0 - c0;
  __syncthreads();

  for (int t = 1; t < Tv; ++t) {
    const float* la = log_a + ((size_t)b * Tv + t) * 64 + j * Kv;
    float mm = -1e30f, vv[8];
    for (int k = 0; k < Kv; ++k) {
      float v = la[k] + alpha[b][k];
      vv[k] = v;
      mm = fmaxf(mm, v);
    }
    float ss = 0.f;
    for (int k = 0; k < Kv; ++k) ss += __expf(vv[k] - mm);
    float av = log_b[((size_t)b * Tv + t) * Kv + j] + mm + __logf(ss);
    __syncthreads();
    tmp[b][j] = av;
    __syncthreads();
    float m2 = -1e30f;
    for (int i = 0; i < Kv; ++i) m2 = fmaxf(m2, tmp[b][i]);
    float s2 = 0.f;
    for (int i = 0; i < Kv; ++i) s2 += __expf(tmp[b][i] - m2);
    float c = m2 + __logf(s2);
    float an = av - c;
    alpha[b][j] = an;
    if (j == 0) lpx[b] += c;
    fwd[(size_t)t * (Bv * Kv) + b * Kv + j] = an;
    __syncthreads();
  }

  alpha[b][j] = 0.f;                                  // beta init
  bwd[(size_t)(Tv - 1) * (Bv * Kv) + b * Kv + j] = 0.f;
  __syncthreads();
  for (int t = Tv - 1; t >= 1; --t) {
    const int k = j;                                  // thread acts as (b,k)
    float mm = -1e30f, vv[8];
    for (int jj = 0; jj < Kv; ++jj) {
      float v = log_a[((size_t)b * Tv + t) * 64 + jj * Kv + k]
              + log_b[((size_t)b * Tv + t) * Kv + jj] + alpha[b][jj];
      vv[jj] = v;
      mm = fmaxf(mm, v);
    }
    float ss = 0.f;
    for (int jj = 0; jj < Kv; ++jj) ss += __expf(vv[jj] - mm);
    float bb = mm + __logf(ss);
    __syncthreads();
    tmp[b][k] = bb;
    __syncthreads();
    float m2 = -1e30f;
    for (int i = 0; i < Kv; ++i) m2 = fmaxf(m2, tmp[b][i]);
    float s2 = 0.f;
    for (int i = 0; i < Kv; ++i) s2 += __expf(tmp[b][i] - m2);
    float bn = bb - (m2 + __logf(s2));
    alpha[b][k] = bn;
    bwd[(size_t)(t - 1) * (Bv * Kv) + b * Kv + k] = bn;
    __syncthreads();
  }
  if (j == 0) log_px[b] = lpx[b];
}

// ---------------------------------------------------------------------------
// Kernel E: emission.  gamma-mix means -> x_gen1 tile -> WMMA @ C (K=32),
// fused Gaussian log-lik vs ys, per-row reduction.
// ---------------------------------------------------------------------------
__global__ __launch_bounds__(256) void emit_kernel(
    const float* __restrict__ ys, const float* __restrict__ x,
    const float* __restrict__ means, const float* __restrict__ fwd,
    const float* __restrict__ bwd, const __bf16* __restrict__ Cb,
    const float* __restrict__ dvec, const float* __restrict__ y_logvar,
    float* __restrict__ log_py) {
  __shared__ alignas(32) __bf16 xF[32][16];
  __shared__ float gw[16][8];
  __shared__ float lpyT[16];

  const int tid = threadIdx.x;
  const int lane = tid & 31, wave = tid >> 5;
  const int bidx = blockIdx.x >> 6;
  const int t0 = (blockIdx.x & 63) * 16;

  if (tid < 16) lpyT[tid] = 0.f;
  if (tid < 128) {
    int row = tid >> 3, k = tid & 7;
    int t = t0 + row;
    gw[row][k] = fwd[(size_t)t * (Bv * Kv) + bidx * Kv + k]
               + bwd[(size_t)t * (Bv * Kv) + bidx * Kv + k];
  }
  __syncthreads();
  if (tid < 16) {
    int row = tid;
    float mx = -1e30f;
    for (int k = 0; k < Kv; ++k) mx = fmaxf(mx, gw[row][k]);
    float s = 0.f;
    for (int k = 0; k < Kv; ++k) s += __expf(gw[row][k] - mx);
    float lse = mx + __logf(s);
    for (int k = 0; k < Kv; ++k) gw[row][k] = __expf(gw[row][k] - lse);
  }
  __syncthreads();

  for (int it = tid; it < 512; it += 256) {
    int row = it >> 5, col = it & 31;
    int t = t0 + row;
    size_t g = (size_t)bidx * Tv + t;
    float xv;
    if (t == 0) {
      xv = x[g * Xv + col];
    } else {
      xv = 0.f;
      for (int k = 0; k < Kv; ++k)
        xv += gw[row][k] * means[((g - 1) * Kv + k) * Xv + col];
    }
    int ln, e;
    afrag_slot(col, row, ln, e);
    xF[ln][e] = (__bf16)xv;
  }
  __syncthreads();

  const int mrow = lane & 15, hi = lane >> 4;
  f32x8 zero8 = {0.f, 0.f, 0.f, 0.f, 0.f, 0.f, 0.f, 0.f};
  bfvec16 a = *(const bfvec16*)&xF[lane][0];
#pragma unroll
  for (int i = 0; i < 4; ++i) {
    int nt = wave * 4 + i;
    bfvec16 bv = *(const bfvec16*)(Cb + ((size_t)nt * 32 + lane) * 16);
    f32x8 acc = wmma_bf16f32(a, bv, zero8);
    int col = nt * 16 + mrow;
    float dv = dvec[col], ylv = y_logvar[col], iv = __expf(-ylv);
#pragma unroll
    for (int r = 0; r < 8; ++r) {
      int row = r + hi * 8;
      size_t g = (size_t)bidx * Tv + t0 + row;
      float diff = ys[g * Dv + col] - (acc[r] + dv);
      atomicAdd(&lpyT[row], -0.5f * (diff * diff * iv + ylv + LOG2PI_F));
    }
  }
  __syncthreads();
  if (tid < 16) log_py[(size_t)bidx * Tv + t0 + tid] = lpyT[tid];
}

// ---------------------------------------------------------------------------
// Kernel F: final reduction.
// elbo = (sum log_px + sum log_py + sum entropy) / B
// ---------------------------------------------------------------------------
__global__ __launch_bounds__(1024) void reduce_kernel(
    const float* __restrict__ log_py, const float* __restrict__ ent,
    const float* __restrict__ log_px, float* __restrict__ out) {
  __shared__ float red[1024];
  int tid = threadIdx.x;
  float s = 0.f;
  for (int i = tid; i < Nv; i += 1024) s += log_py[i] + ent[i];
  if (tid < Bv) s += log_px[tid];
  red[tid] = s;
  __syncthreads();
  for (int st = 512; st > 0; st >>= 1) {
    if (tid < st) red[tid] += red[tid + st];
    __syncthreads();
  }
  if (tid == 0) out[0] = red[0] / (float)Bv;
}

// ---------------------------------------------------------------------------
extern "C" void kernel_launch(void* const* d_in, const int* in_sizes, int n_in,
                              void* d_out, int out_size, void* d_ws,
                              size_t ws_size, hipStream_t stream) {
  const float* ys        = (const float*)d_in[0];
  const float* us        = (const float*)d_in[1];
  const float* eps       = (const float*)d_in[2];
  const float* W1        = (const float*)d_in[3];
  const float* b1        = (const float*)d_in[4];
  const float* Wmu       = (const float*)d_in[5];
  const float* bmu       = (const float*)d_in[6];
  const float* Wlv       = (const float*)d_in[7];
  const float* blv       = (const float*)d_in[8];
  const float* Wz        = (const float*)d_in[9];
  const float* bz        = (const float*)d_in[10];
  const float* A         = (const float*)d_in[11];
  const float* Bu        = (const float*)d_in[12];
  const float* bx        = (const float*)d_in[13];
  const float* x_logvar  = (const float*)d_in[14];
  const float* x0_mean   = (const float*)d_in[15];
  const float* x0_logvar = (const float*)d_in[16];
  const float* C         = (const float*)d_in[17];
  const float* dvec      = (const float*)d_in[18];
  const float* y_logvar  = (const float*)d_in[19];
  const float* log_initz = (const float*)d_in[20];

  char* ws = (char*)d_ws;
  __bf16* W1b   = (__bf16*)(ws + OFF_W1B);
  __bf16* Wmub  = (__bf16*)(ws + OFF_WMUB);
  __bf16* Wlvb  = (__bf16*)(ws + OFF_WLVB);
  __bf16* Wzb   = (__bf16*)(ws + OFF_WZB);
  __bf16* Cb    = (__bf16*)(ws + OFF_CB);
  __bf16* Ab    = (__bf16*)(ws + OFF_AB);
  float* xbuf   = (float*)(ws + OFF_X);
  float* entb   = (float*)(ws + OFF_ENT);
  float* log_a  = (float*)(ws + OFF_LOGA);
  float* log_b  = (float*)(ws + OFF_LOGB);
  float* means  = (float*)(ws + OFF_MEANS);
  float* fwd    = (float*)(ws + OFF_FWD);
  float* bwd    = (float*)(ws + OFF_BWD);
  float* lpx    = (float*)(ws + OFF_LPX);
  float* lpy    = (float*)(ws + OFF_LPY);

  auto pack = [&](const float* W, __bf16* P, int Kd, int Nd, int nMat) {
    int total = (Nd / 16) * (Kd / 32) * 512 * nMat;
    pack_b_kernel<<<(total + 255) / 256, 256, 0, stream>>>(W, P, Kd, Nd, nMat);
  };
  pack(W1,  W1b,  Dv, Hv, 1);
  pack(Wmu, Wmub, Dv, Xv, 1);
  pack(Wlv, Wlvb, Dv, Xv, 1);
  pack(Wz,  Wzb,  Xv, Kv * Kv, 1);
  pack(C,   Cb,   Xv, Dv, 1);
  pack(A,   Ab,   Xv, Xv, Kv);

  infer_kernel<<<Nv / 16, 256, 0, stream>>>(ys, eps, W1b, b1, Wmub, bmu,
                                            Wlvb, blv, xbuf, entb);
  dyn_kernel<<<Bv * (Tv / 16), 256, 0, stream>>>(xbuf, us, Wzb, bz, Ab, Bu,
                                                 bx, x_logvar, log_a, log_b,
                                                 means);
  logb0_kernel<<<1, 256, 0, stream>>>(xbuf, x0_mean, x0_logvar, log_b);
  fb_kernel<<<1, 256, 0, stream>>>(log_a, log_b, log_initz, fwd, bwd, lpx);
  emit_kernel<<<Bv * (Tv / 16), 256, 0, stream>>>(ys, xbuf, means, fwd, bwd,
                                                  Cb, dvec, y_logvar, lpy);
  reduce_kernel<<<1, 1024, 0, stream>>>(lpy, entb, lpx, (float*)d_out);
}
